// NSVQ_20744692040084
// MI455X (gfx1250) — matrix operand order repaired and checked by reading, not compiled, not run on hardware
//
#include <hip/hip_runtime.h>

typedef __attribute__((ext_vector_type(16))) __bf16          v16bf;
typedef __attribute__((ext_vector_type(16))) unsigned short  v16u;
typedef __attribute__((ext_vector_type(8)))  float           v8f;

#define D_DIM 64
#define SCORE_BIAS 32.0f

__device__ __forceinline__ unsigned short f2bf(float f) {
    // round-to-nearest-even fp32 -> bf16
    unsigned int u = __float_as_uint(f);
    u += 0x7FFFu + ((u >> 16) & 1u);
    return (unsigned short)(u >> 16);
}

// ---------------------------------------------------------------------------
// K0: convert codebooks to bf16, compute ||c||^2 (fp32), zero histogram.
// grid: K/8 blocks x 256 threads (8 waves, one codebook row per wave)
// ---------------------------------------------------------------------------
__global__ __launch_bounds__(256) void nsvq_prep(
    const float* __restrict__ cb, unsigned short* __restrict__ cb_bf,
    float* __restrict__ cnorm, int* __restrict__ counts, int K)
{
    int lane = threadIdx.x & 31;
    int wave = threadIdx.x >> 5;
    int tid  = blockIdx.x * 256 + threadIdx.x;
    if (tid < K) counts[tid] = 0;

    int k = blockIdx.x * 8 + wave;
    const float* cr = cb + (long)k * D_DIM;
    float c0 = cr[lane], c1 = cr[lane + 32];
    cb_bf[(long)k * D_DIM + lane]      = f2bf(c0);
    cb_bf[(long)k * D_DIM + 32 + lane] = f2bf(c1);
    float s = c0 * c0 + c1 * c1;
    #pragma unroll
    for (int m = 16; m >= 1; m >>= 1) s += __shfl_xor(s, m, 32);
    if (lane == 0) cnorm[k] = s;
}

// ---------------------------------------------------------------------------
// K1: bf16 WMMA GEMM (X @ C^T) fused with argmin over K.
// 256 threads = 8 waves; each wave owns 32 rows (two 16-row A tiles that
// share every B load -> 4 WMMAs per ds_load pair). Whole bf16 codebook
// (128 KB) + biased ||c||^2 (4 KB) staged in LDS, reused by 256 rows/block.
// Argmin: 10-bit index packed into score mantissa -> single v_min_i32 chain.
// ---------------------------------------------------------------------------
__global__ __launch_bounds__(256) void nsvq_argmin(
    const float* __restrict__ X, const unsigned short* __restrict__ cb_bf,
    const float* __restrict__ cnorm, int* __restrict__ min_idx,
    int* __restrict__ counts, int K)
{
    extern __shared__ unsigned short smem[];
    unsigned short* s_cb = smem;                               // K*64 bf16
    float*          s_cn = (float*)(smem + (size_t)K * D_DIM); // K floats (biased)

    // cooperative stage: codebook (uint4 chunks) + biased cnorm
    {
        const uint4* src = (const uint4*)cb_bf;
        uint4*       dst = (uint4*)s_cb;
        int n16 = K * D_DIM / 8;                               // #uint4 chunks
        for (int i = threadIdx.x; i < n16; i += 256) dst[i] = src[i];
        for (int i = threadIdx.x; i < K;   i += 256) s_cn[i] = cnorm[i] + SCORE_BIAS;
    }
    __syncthreads();

    int lane = threadIdx.x & 31;
    int wave = threadIdx.x >> 5;
    int r15  = lane & 15;
    int hi   = lane >> 4;                                      // half-wave
    long rowBase = ((long)blockIdx.x * 8 + wave) * 32;
    const float* xr0 = X + (rowBase + r15) * D_DIM;
    const float* xr1 = X + (rowBase + 16 + r15) * D_DIM;

    // A operands: contiguous k order chosen by this code (same as B).
    // lanes 0-15: K = {0..15}; lanes 16-31: K = {16..31}
    v16u a0u, a1u, a2u, a3u;
    #pragma unroll
    for (int i = 0; i < 8; ++i) {
        a0u[i]     = f2bf(xr0[16 * hi + i]);
        a0u[i + 8] = f2bf(xr0[8 + 16 * hi + i]);
        a1u[i]     = f2bf(xr0[32 + 16 * hi + i]);
        a1u[i + 8] = f2bf(xr0[40 + 16 * hi + i]);
        a2u[i]     = f2bf(xr1[16 * hi + i]);
        a2u[i + 8] = f2bf(xr1[8 + 16 * hi + i]);
        a3u[i]     = f2bf(xr1[32 + 16 * hi + i]);
        a3u[i + 8] = f2bf(xr1[40 + 16 * hi + i]);
    }
    v16bf a0 = __builtin_bit_cast(v16bf, a0u);
    v16bf a1 = __builtin_bit_cast(v16bf, a1u);
    v16bf a2 = __builtin_bit_cast(v16bf, a2u);
    v16bf a3 = __builtin_bit_cast(v16bf, a3u);

    // packed (score<<10 | index) minima; scores guaranteed > 0 via SCORE_BIAS
    int best[16];
    #pragma unroll
    for (int r = 0; r < 16; ++r) best[r] = 0x7FFFFFFF;

    int nTiles = K / 16;
    // B operands: 16-bit B 32x16 layout, lane = column, dims contiguous.
    // Column stride between tiles = 16 cols * 64 dims = 1024 ushorts.
    const unsigned short* bbase = s_cb + (size_t)r15 * D_DIM + 16 * hi;
    v16u b0u = *(const v16u*)(bbase);
    v16u b1u = *(const v16u*)(bbase + 32);

    #pragma unroll 2
    for (int kt = 0; kt < nTiles; ++kt) {
        v16bf b0 = __builtin_bit_cast(v16bf, b0u);
        v16bf b1 = __builtin_bit_cast(v16bf, b1u);
        v8f acc0 = {}, acc1 = {};
        acc0 = __builtin_amdgcn_wmma_f32_16x16x32_bf16(false, a0, false, b0,
                                                       (short)0, acc0, false, false);
        acc1 = __builtin_amdgcn_wmma_f32_16x16x32_bf16(false, a2, false, b0,
                                                       (short)0, acc1, false, false);
        acc0 = __builtin_amdgcn_wmma_f32_16x16x32_bf16(false, a1, false, b1,
                                                       (short)0, acc0, false, false);
        acc1 = __builtin_amdgcn_wmma_f32_16x16x32_bf16(false, a3, false, b1,
                                                       (short)0, acc1, false, false);

        int   nb = kt * 16 + r15;          // this lane's column index
        float cn = s_cn[nb];               // ||c||^2 + bias

        // software-pipeline: prefetch next tile's B while argmin VALU runs
        if (kt + 1 < nTiles) {
            const unsigned short* bn = bbase + (size_t)(kt + 1) * 1024;
            b0u = *(const v16u*)(bn);
            b1u = *(const v16u*)(bn + 32);
        }

        #pragma unroll
        for (int r = 0; r < 8; ++r) {
            // rank by ||c||^2 - 2*sim (+bias): ||x||^2 constant per row
            float s0 = __builtin_fmaf(-2.0f, acc0[r], cn);
            float s1 = __builtin_fmaf(-2.0f, acc1[r], cn);
            int p0 = (__float_as_int(s0) & ~1023) | nb;   // v_and_or_b32
            int p1 = (__float_as_int(s1) & ~1023) | nb;
            best[r]     = min(best[r],     p0);           // v_min_i32
            best[r + 8] = min(best[r + 8], p1);
        }
    }

    // reduce over the 16 columns: xor masks 1,2,4,8 stay inside each half-wave
    #pragma unroll
    for (int m = 1; m < 16; m <<= 1) {
        #pragma unroll
        for (int r = 0; r < 16; ++r) {
            int op = __shfl_xor(best[r], m, 32);
            best[r] = min(best[r], op);
        }
    }
    // C/D layout: slot r holds row r (lanes 0-15) / row 8+r (lanes 16-31)
    if (r15 == 0) {
        long rb = rowBase + 8 * hi;
        #pragma unroll
        for (int r = 0; r < 8; ++r) {
            int bi = best[r] & 1023;
            min_idx[rb + r] = bi;
            atomicAdd(&counts[bi], 1);
        }
        rb += 16;
        #pragma unroll
        for (int r = 0; r < 8; ++r) {
            int bi = best[r + 8] & 1023;
            min_idx[rb + r] = bi;
            atomicAdd(&counts[bi], 1);
        }
    }
}

// ---------------------------------------------------------------------------
// K2: noise substitution, exact fp32. One wave per row, 2 dims per lane.
// ---------------------------------------------------------------------------
__global__ __launch_bounds__(256) void nsvq_finalize(
    const float* __restrict__ X, const float* __restrict__ RV,
    const float* __restrict__ CB, const int* __restrict__ min_idx,
    float* __restrict__ out)
{
    int lane = threadIdx.x & 31;
    int wave = threadIdx.x >> 5;
    long row = (long)blockIdx.x * 8 + wave;
    int  idx = min_idx[row];
    const float* xr = X  + row * D_DIM;
    const float* rr = RV + row * D_DIM;
    const float* cr = CB + (long)idx * D_DIM;
    float x0 = xr[lane], x1 = xr[lane + 32];
    float r0 = rr[lane], r1 = rr[lane + 32];
    float c0 = cr[lane], c1 = cr[lane + 32];
    float d0 = x0 - c0, d1 = x1 - c1;
    float sres = d0 * d0 + d1 * d1;
    float srv  = r0 * r0 + r1 * r1;
    #pragma unroll
    for (int m = 16; m >= 1; m >>= 1) {
        sres += __shfl_xor(sres, m, 32);
        srv  += __shfl_xor(srv,  m, 32);
    }
    float scale = __builtin_sqrtf(sres) / (__builtin_sqrtf(srv) + 1e-12f);
    float* o = out + row * D_DIM;
    o[lane]      = __builtin_fmaf(scale, r0, x0);
    o[lane + 32] = __builtin_fmaf(scale, r1, x1);
}

// ---------------------------------------------------------------------------
// K3: perplexity + usage counters. One block, K threads.
// ---------------------------------------------------------------------------
__global__ void nsvq_stats(const int* __restrict__ counts,
                           const int* __restrict__ used,
                           float* __restrict__ out_perp,
                           int* __restrict__ out_used, float invN, int K)
{
    __shared__ float red[1024];
    int t = threadIdx.x;
    int c = counts[t];
    out_used[t] = used[t] + c;
    float avg = (float)c * invN;
    red[t] = avg * logf(avg + 1e-12f);
    __syncthreads();
    for (int s = K >> 1; s > 0; s >>= 1) {
        if (t < s) red[t] += red[t + s];
        __syncthreads();
    }
    if (t == 0) *out_perp = expf(-red[0]);
}

// ---------------------------------------------------------------------------
extern "C" void kernel_launch(void* const* d_in, const int* in_sizes, int n_in,
                              void* d_out, int out_size, void* d_ws, size_t ws_size,
                              hipStream_t stream)
{
    const float* X    = (const float*)d_in[0];   // [N, 64]
    const float* RV   = (const float*)d_in[1];   // [N, 64]
    const float* CB   = (const float*)d_in[2];   // [K, 64]
    const int*   used = (const int*)  d_in[3];   // [K]

    const int N = in_sizes[0] / D_DIM;           // 262144
    const int K = in_sizes[2] / D_DIM;           // 1024

    float* out      = (float*)d_out;             // [N*64] quantized
    size_t ND       = (size_t)N * D_DIM;
    float* out_perp = out + ND;                  // [1]
    int*   out_used = (int*)d_out + ND + 1;      // [K] int32 bits

    // workspace layout
    char* ws = (char*)d_ws;
    unsigned short* cb_bf  = (unsigned short*)ws;                       // K*64 bf16
    float*          cnorm  = (float*)(ws + (size_t)K * D_DIM * 2);      // K f32
    int*            counts = (int*)  (ws + (size_t)K * D_DIM * 2 + (size_t)K * 4);
    int*            minidx = (int*)  (ws + (size_t)K * D_DIM * 2 + (size_t)K * 8);

    nsvq_prep<<<K / 8, 256, 0, stream>>>(CB, cb_bf, cnorm, counts, K);

    size_t lds = (size_t)K * D_DIM * 2 + (size_t)K * 4;   // 128KB cb + 4KB cnorm
    nsvq_argmin<<<N / 256, 256, lds, stream>>>(X, cb_bf, cnorm, minidx, counts, K);

    nsvq_finalize<<<N / 8, 256, 0, stream>>>(X, RV, CB, minidx, out);

    nsvq_stats<<<1, K, 0, stream>>>(counts, used, out_perp, out_used,
                                    1.0f / (float)N, K);
}
